// MyQuantConv2d_65360812311363
// MI455X (gfx1250) — 8-bit matrix operand order repaired and checked by reading, compiled, not run on hardware
//
#include <hip/hip_runtime.h>
#include <math.h>

#define BATCH 16
#define IN_CH 64
#define OUT_CH 128
#define HW 128
#define TAPS 9
#define KSIZE (IN_CH * TAPS)
#define NTILES (OUT_CH / 16)

typedef int v8i __attribute__((ext_vector_type(8)));

#define AS1 __attribute__((address_space(1)))
#define AS3 __attribute__((address_space(3)))

#if defined(__has_builtin)
#if __has_builtin(__builtin_amdgcn_global_load_async_to_lds_b32) && \
    __has_builtin(__builtin_amdgcn_s_wait_asynccnt)
#define USE_ASYNC_STAGE 1
#endif
#endif

#define QMAGIC 12582912.f
__device__ __forceinline__ unsigned qbits(float x, float s) {
  float v = fminf(fmaxf(x * s, -127.f), 127.f);
  v += QMAGIC;
  return __float_as_uint(v);
}
__device__ __forceinline__ unsigned pack_lo8(unsigned f0, unsigned f1,
                                             unsigned f2, unsigned f3) {
  unsigned t01 = __builtin_amdgcn_perm(f1, f0, 0x0c0c0400u);
  unsigned t23 = __builtin_amdgcn_perm(f3, f2, 0x04000c0cu);
  return t01 | t23;
}

__global__ void k_zero(unsigned* __restrict__ p, int n) {
  int i = blockIdx.x * blockDim.x + threadIdx.x;
  if (i < n) p[i] = 0u;
}

__global__ void k_act_amax(const float* __restrict__ in, unsigned* __restrict__ amax) {
  __shared__ float red[256];
  int bc = blockIdx.x;
  int c  = bc & (IN_CH - 1);
  const float* p = in + (size_t)bc * (HW * HW);
  float m[TAPS];
#pragma unroll
  for (int t = 0; t < TAPS; t++) m[t] = 0.f;
  for (int i = threadIdx.x; i < HW * HW; i += 256) {
    int h = i >> 7, w = i & (HW - 1);
    float a = fabsf(p[i]);
    bool vh[3] = { h < HW - 1, true, h > 0 };
    bool vw[3] = { w < HW - 1, true, w > 0 };
#pragma unroll
    for (int kh = 0; kh < 3; kh++)
#pragma unroll
      for (int kw = 0; kw < 3; kw++)
        m[kh * 3 + kw] = fmaxf(m[kh * 3 + kw], (vh[kh] && vw[kw]) ? a : 0.f);
  }
  for (int t = 0; t < TAPS; t++) {
    red[threadIdx.x] = m[t];
    __syncthreads();
    for (int s = 128; s > 0; s >>= 1) {
      if (threadIdx.x < s) red[threadIdx.x] = fmaxf(red[threadIdx.x], red[threadIdx.x + s]);
      __syncthreads();
    }
    if (threadIdx.x == 0)
      atomicMax(&amax[c * TAPS + t], __float_as_uint(red[0]));
    __syncthreads();
  }
}

__global__ void k_scales(const float* __restrict__ wt, const unsigned* __restrict__ amax,
                         float* __restrict__ inv_x, float* __restrict__ inv_w,
                         float* __restrict__ s_out) {
  __shared__ float rx[KSIZE], rw[KSIZE];
  int k = threadIdx.x;
  float wmax = 0.f;
  for (int o = 0; o < OUT_CH; o++) wmax = fmaxf(wmax, fabsf(wt[o * KSIZE + k]));
  float amx = __uint_as_float(amax[k]);
  float scale = sqrtf(amx) / sqrtf(wmax);
  if (scale == 0.f) scale = 1.f;
  rx[k] = amx / scale;
  rw[k] = wmax * scale;
  __syncthreads();
  for (int s = 512; s > 0; s >>= 1) {
    if (k < s && k + s < KSIZE) {
      rx[k] = fmaxf(rx[k], rx[k + s]);
      rw[k] = fmaxf(rw[k], rw[k + s]);
    }
    __syncthreads();
  }
  float sx = rx[0] > 0.f ? rx[0] / 127.f : 1.f;
  float sw = rw[0] > 0.f ? rw[0] / 127.f : 1.f;
  inv_x[k] = 1.f / (scale * sx);
  inv_w[k] = scale / sw;
  if (k == 0) s_out[0] = sx * sw;
}

__global__ void k_packw(const float* __restrict__ wt, const float* __restrict__ inv_w,
                        unsigned* __restrict__ wpack) {
  int idx = blockIdx.x * 256 + threadIdx.x;
  if (idx >= NTILES * TAPS * 32 * 8) return;
  int v = idx & 7;
  int lane = (idx >> 3) & 31;
  int g = idx >> 8;
  int nt = g / TAPS, t = g - nt * TAPS;
  int oc = nt * 16 + (lane & 15);
  int c0 = ((v >> 2) << 5) + ((lane >> 4) << 4) + (v & 3) * 4;
  unsigned f[4];
#pragma unroll
  for (int bb = 0; bb < 4; bb++) {
    int kg = (c0 + bb) * TAPS + t;
    f[bb] = qbits(wt[oc * KSIZE + kg], inv_w[kg]);
  }
  wpack[idx] = pack_lo8(f[0], f[1], f[2], f[3]);
}

#define IN_STRIDE 68
#define INVX_OFF  13464

__global__ __launch_bounds__(256) void k_conv(const float* __restrict__ in,
                                              const unsigned* __restrict__ wpack,
                                              const float* __restrict__ inv_x,
                                              const float* __restrict__ s_out,
                                              float* __restrict__ out) {
  __shared__ __align__(16) float smem[INVX_OFF + KSIZE];
  int tid = threadIdx.x;
  int wg = blockIdx.x;
  int ow0 = (wg & 1) * 64;
  int oh = (wg >> 1) & (HW - 1);
  int b = wg >> 8;

  for (int i = tid; i < KSIZE; i += 256) {
    int t = i >> 6, c = i & 63;
    smem[INVX_OFF + i] = inv_x[c * TAPS + t];
  }
  for (int i = tid; i < 3 * 66 * 64; i += 256) {
    int col = i % 66;
    int rc = i / 66;
    int r = rc % 3;
    int c = rc / 3;
    int h = oh - 1 + r;
    int w = ow0 - 1 + col;
    float* dst = &smem[(r * 66 + col) * IN_STRIDE + c];
    if (h >= 0 && h < HW && w >= 0 && w < HW) {
      const float* src = &in[(((size_t)b * IN_CH + c) * HW + h) * HW + w];
#ifdef USE_ASYNC_STAGE
      __builtin_amdgcn_global_load_async_to_lds_b32(
          (AS1 int*)src, (AS3 int*)dst, 0, 0);
#else
      *dst = *src;
#endif
    } else {
      *dst = 0.f;
    }
  }
#ifdef USE_ASYNC_STAGE
  __builtin_amdgcn_s_wait_asynccnt(0);
#endif
  __syncthreads();

  int lane = tid & 31;
  int wv = tid >> 5;
  int msub = wv & 3;
  int ng = wv >> 2;
  int half = lane >> 4;
  int mrow = lane & 15;

  v8i acc[4] = {};
  const float* invx = &smem[INVX_OFF];

#pragma unroll
  for (int kh = 0; kh < 3; kh++) {
#pragma unroll
    for (int kw = 0; kw < 3; kw++) {
      int t = kh * 3 + kw;
      int col = msub * 16 + mrow + kw;
      v8i a = {};
#pragma unroll
      for (int v = 0; v < 8; v++) {
        int c0 = ((v >> 2) << 5) + half * 16 + ((v & 3) << 2);
        const float4 x4 = *(const float4*)&smem[(kh * 66 + col) * IN_STRIDE + c0];
        const float4 s4 = *(const float4*)&invx[t * 64 + c0];
        a[v] = (int)pack_lo8(qbits(x4.x, s4.x), qbits(x4.y, s4.y),
                             qbits(x4.z, s4.z), qbits(x4.w, s4.w));
      }
#pragma unroll
      for (int j = 0; j < 4; j++) {
        int nt = ng * 4 + j;
        const int4* wp = (const int4*)&wpack[(((nt * TAPS + t) * 32 + lane) << 3)];
        int4 b0 = wp[0], b1 = wp[1];
        v8i bf;
        bf[0] = b0.x; bf[1] = b0.y; bf[2] = b0.z; bf[3] = b0.w;
        bf[4] = b1.x; bf[5] = b1.y; bf[6] = b1.z; bf[7] = b1.w;
        acc[j] = __builtin_amdgcn_wmma_i32_16x16x64_iu8(
             true, a,  true, bf, acc[j],
             false,  false);
      }
    }
  }
  __syncthreads();
  float so = s_out[0];
#pragma unroll
  for (int j = 0; j < 4; j++) {
    int oc = (ng * 4 + j) * 16 + mrow;
#pragma unroll
    for (int v = 0; v < 8; v++)
      smem[oc * 64 + msub * 16 + half * 8 + v] = (float)acc[j][v] * so;
  }
  __syncthreads();
  for (int i = tid; i < OUT_CH * 64; i += 256) {
    int oc = i >> 6, wl = i & 63;
    out[(((size_t)b * OUT_CH + oc) * HW + oh) * HW + ow0 + wl] = smem[i];
  }
}

extern "C" void kernel_launch(void* const* d_in, const int* in_sizes, int n_in,
                              void* d_out, int out_size, void* d_ws, size_t ws_size,
                              hipStream_t stream) {
  const float* in = (const float*)d_in[0];
  const float* wt = (const float*)d_in[1];
  float* out = (float*)d_out;
  char* ws = (char*)d_ws;

  unsigned* amax  = (unsigned*)(ws + 0);
  float*    inv_x = (float*)(ws + 2304);
  float*    inv_w = (float*)(ws + 4608);
  float*    s_o   = (float*)(ws + 6912);
  unsigned* wpack = (unsigned*)(ws + 7168);

  k_zero<<<3, 256, 0, stream>>>(amax, KSIZE);
  k_act_amax<<<BATCH * IN_CH, 256, 0, stream>>>(in, amax);
  k_scales<<<1, KSIZE, 0, stream>>>(wt, amax, inv_x, inv_w, s_o);
  k_packw<<<72, 256, 0, stream>>>(wt, inv_w, wpack);
  k_conv<<<BATCH * HW * 2, 256, 0, stream>>>(in, wpack, inv_x, s_o, out);
}
